// GNN_QNetwork_19378892439726
// MI455X (gfx1250) — matrix operand order repaired and checked by reading, not compiled, not run on hardware
//
#include <hip/hip_runtime.h>
#include <hip/hip_bf16.h>

typedef __attribute__((ext_vector_type(16))) _Float16 v16h;
typedef __attribute__((ext_vector_type(8)))  float    v8f;

#define HID   128
#define HEADS 4
#define PH    32
#define SLOPE 0.2f

// ---------------------------------------------------------------------------
// Weight pre-pack: B[K,nout] f32 (row-major)  ->  f16 WMMA B-fragments.
// Fragment f = ks*CT + ct is 32 lanes x 16 halfs (one v16h per lane),
// laid out per CDNA5 ISA 7.12.2: lanes 0-15 hold K=ks*32+{0..7,16..23} (2 per slot),
// lanes 16-31 hold K=ks*32+{8..15,24..31}; N = ct*16 + (lane&15).
// ---------------------------------------------------------------------------
__global__ __launch_bounds__(256) void pack_b_k(
    const float* __restrict__ B, _Float16* __restrict__ Bp, int K, int nout)
{
    const int t = blockIdx.x * blockDim.x + threadIdx.x;
    const int CT = nout >> 4;
    const int total = (K >> 5) * CT * 32;
    if (t >= total) return;
    const int lane = t & 31, frag = t >> 5;
    const int ks = frag / CT, ct = frag - ks * CT;
    const int sel = lane >> 4, l = lane & 15;
    const int n = ct * 16 + l;
    _Float16* dst = Bp + (size_t)t * 16;
#pragma unroll
    for (int j = 0; j < 8; ++j) {
        const int kb = ks * 32 + sel * 8 + 2 * j + (j >= 4 ? 8 : 0);
        dst[2 * j]     = (_Float16)B[(size_t)kb * nout + n];
        dst[2 * j + 1] = (_Float16)B[(size_t)(kb + 1) * nout + n];
    }
}

// ---------------------------------------------------------------------------
// WMMA GEMM: C[M,NOUT] = act(A[M,K] @ B + bias), B pre-packed f16 fragments.
// One wave -> 16 rows x NOUT cols: A fragment loaded once per k-step and
// reused across all column tiles (A read exactly once per GEMM).
// Fully unrolled: base-register + immediate-offset loads, static WMMA chain.
// ---------------------------------------------------------------------------
template <int K, int NOUT, int RELU>
__global__ __launch_bounds__(256) void gemm_wmma(
    const float* __restrict__ A, const _Float16* __restrict__ Bp,
    const float* __restrict__ bias, float* __restrict__ C, int M)
{
    constexpr int CT = NOUT / 16;   // column tiles
    constexpr int KS = K / 32;      // k-steps

    const int rowTile = blockIdx.x * (blockDim.x >> 5) + (threadIdx.x >> 5);
    if (rowTile * 16 >= M) return;
    const int lane = threadIdx.x & 31;
    const int sel = lane >> 4, l = lane & 15;

    v8f acc[CT];
#pragma unroll
    for (int ct = 0; ct < CT; ++ct) acc[ct] = (v8f){};

    // A layout (16x32 MxK): lane half sel -> K runs {sel*8..+7, 16+sel*8..+7}
    const float* Ab = A + (size_t)(rowTile * 16 + l) * K + sel * 8;
    const v16h*  Bb = (const v16h*)Bp + lane;   // fragment f at Bb[f*32]

#pragma unroll
    for (int ks = 0; ks < KS; ++ks) {
        const float4 a0 = *(const float4*)(Ab + ks * 32 + 0);
        const float4 a1 = *(const float4*)(Ab + ks * 32 + 4);
        const float4 a2 = *(const float4*)(Ab + ks * 32 + 16);
        const float4 a3 = *(const float4*)(Ab + ks * 32 + 20);
        v16h av;
        av[0]  = (_Float16)a0.x; av[1]  = (_Float16)a0.y;
        av[2]  = (_Float16)a0.z; av[3]  = (_Float16)a0.w;
        av[4]  = (_Float16)a1.x; av[5]  = (_Float16)a1.y;
        av[6]  = (_Float16)a1.z; av[7]  = (_Float16)a1.w;
        av[8]  = (_Float16)a2.x; av[9]  = (_Float16)a2.y;
        av[10] = (_Float16)a2.z; av[11] = (_Float16)a2.w;
        av[12] = (_Float16)a3.x; av[13] = (_Float16)a3.y;
        av[14] = (_Float16)a3.z; av[15] = (_Float16)a3.w;
#pragma unroll
        for (int ct = 0; ct < CT; ++ct) {
            const v16h bv = Bb[(ks * CT + ct) * 32];
            acc[ct] = __builtin_amdgcn_wmma_f32_16x16x32_f16(
                /*neg_a=*/false, av, /*neg_b=*/false, bv,
                /*c_mod=*/(short)0, acc[ct], /*reuse_a=*/false, /*reuse_b=*/false);
        }
    }

    // C layout: VGPR r -> M = r + sel*8, N = ct*16 + l
    float* Cb = C + (size_t)(rowTile * 16 + sel * 8) * NOUT + l;
#pragma unroll
    for (int ct = 0; ct < CT; ++ct) {
        const float bb = bias[ct * 16 + l];
#pragma unroll
        for (int r = 0; r < 8; ++r) {
            float v = acc[ct][r] + bb;
            if (RELU) v = fmaxf(v, 0.f);
            Cb[(size_t)r * NOUT + ct * 16] = v;
        }
    }
}

// ---------------------------------------------------------------------------
// Per-node attention logits: a_src[n,h] = <h_proj[n,h,:], att_src[h,:]>
// ---------------------------------------------------------------------------
__global__ __launch_bounds__(128) void att_scores_k(
    const float* __restrict__ h, const float* __restrict__ att_src,
    const float* __restrict__ att_dst, float* __restrict__ a_src,
    float* __restrict__ a_dst)
{
    const int nidx = blockIdx.x;
    const int i    = threadIdx.x;
    const int head = i >> 5, c = i & 31;
    float v  = h[(size_t)nidx * HID + i];
    float ps = v * att_src[head * PH + c];
    float pd = v * att_dst[head * PH + c];
#pragma unroll
    for (int off = 16; off > 0; off >>= 1) {
        ps += __shfl_xor(ps, off, 32);
        pd += __shfl_xor(pd, off, 32);
    }
    if (c == 0) {
        a_src[nidx * HEADS + head] = ps;
        a_dst[nidx * HEADS + head] = pd;
    }
}

// Order-preserving float<->uint encoding for atomic max.
__device__ __forceinline__ unsigned enc_f32(float f) {
    unsigned u = __float_as_uint(f);
    return u ^ ((u >> 31) ? 0xFFFFFFFFu : 0x80000000u);
}
__device__ __forceinline__ float dec_f32(unsigned u) {
    unsigned b = (u & 0x80000000u) ? (u ^ 0x80000000u) : ~u;
    return __uint_as_float(b);
}
__device__ __forceinline__ float lrelu(float e) { return e > 0.f ? e : SLOPE * e; }

__device__ __forceinline__ void edge_ends(const long long* ei, long long t,
                                          int E, int& s, int& d) {
    if (t < (long long)E) { s = (int)ei[t]; d = (int)ei[(long long)E + t]; }
    else                  { s = d = (int)(t - E); }   // self-loop
}

// Pass 1: per-(dst,head) running max of leaky_relu(a_src[src]+a_dst[dst]).
__global__ __launch_bounds__(256) void edge_max_k(
    const long long* __restrict__ ei, int E, int Nn,
    const float* __restrict__ a_src, const float* __restrict__ a_dst,
    unsigned* __restrict__ mval)
{
    long long t = (long long)blockIdx.x * blockDim.x + threadIdx.x;
    if (t >= (long long)E + Nn) return;
    int s, d; edge_ends(ei, t, E, s, d);
    const float4 as = *(const float4*)(a_src + (size_t)s * HEADS);
    const float4 ad = *(const float4*)(a_dst + (size_t)d * HEADS);
    atomicMax(&mval[d * HEADS + 0], enc_f32(lrelu(as.x + ad.x)));
    atomicMax(&mval[d * HEADS + 1], enc_f32(lrelu(as.y + ad.y)));
    atomicMax(&mval[d * HEADS + 2], enc_f32(lrelu(as.z + ad.z)));
    atomicMax(&mval[d * HEADS + 3], enc_f32(lrelu(as.w + ad.w)));
}

// Pass 2: ex = exp(e - max); stash per-edge ex, accumulate denom per dst.
__global__ __launch_bounds__(256) void edge_expsum_k(
    const long long* __restrict__ ei, int E, int Nn,
    const float* __restrict__ a_src, const float* __restrict__ a_dst,
    const unsigned* __restrict__ mval, float* __restrict__ exw,
    float* __restrict__ denom)
{
    long long t = (long long)blockIdx.x * blockDim.x + threadIdx.x;
    if (t >= (long long)E + Nn) return;
    int s, d; edge_ends(ei, t, E, s, d);
    const float4 as = *(const float4*)(a_src + (size_t)s * HEADS);
    const float4 ad = *(const float4*)(a_dst + (size_t)d * HEADS);
    const float ev[4] = { as.x + ad.x, as.y + ad.y, as.z + ad.z, as.w + ad.w };
#pragma unroll
    for (int h = 0; h < HEADS; ++h) {
        float ex = __expf(lrelu(ev[h]) - dec_f32(mval[d * HEADS + h]));
        exw[(size_t)t * HEADS + h] = ex;
        atomicAdd(&denom[d * HEADS + h], ex);
    }
}

// Pass 3: out[dst,:] += alpha * h_proj[src,:].  Thread per (edge, channel):
// coalesced 512B src-row reads, native f32 global atomics for the scatter.
__global__ __launch_bounds__(256) void edge_agg_k(
    const long long* __restrict__ ei, int E, int Nn,
    const float* __restrict__ hsrc, const float* __restrict__ exw,
    const float* __restrict__ denom, float* __restrict__ out)
{
    size_t t = (size_t)blockIdx.x * blockDim.x + threadIdx.x;
    size_t total = ((size_t)E + (size_t)Nn) * HID;
    if (t >= total) return;
    long long e = (long long)(t >> 7);
    int i = (int)(t & (HID - 1));
    int h = i >> 5;
    int s, d; edge_ends(ei, e, E, s, d);
    float alpha = exw[(size_t)e * HEADS + h] / denom[(size_t)d * HEADS + h];
    atomicAdd(out + (size_t)d * HID + i, alpha * hsrc[(size_t)s * HID + i]);
}

// out[n,i] = bias[i]  (seeds the aggregation buffer so pass 3 lands on bias)
__global__ __launch_bounds__(256) void seed_bias_k(
    float* __restrict__ out, const float* __restrict__ bias, size_t total)
{
    size_t t = (size_t)blockIdx.x * blockDim.x + threadIdx.x;
    if (t < total) out[t] = bias[t & (HID - 1)];
}

__global__ __launch_bounds__(256) void relu_inplace_k(
    float* __restrict__ x, size_t total)
{
    size_t t = (size_t)blockIdx.x * blockDim.x + threadIdx.x;
    if (t < total) x[t] = fmaxf(x[t], 0.f);
}

// ---------------------------------------------------------------------------
extern "C" void kernel_launch(void* const* d_in, const int* in_sizes, int n_in,
                              void* d_out, int out_size, void* d_ws, size_t ws_size,
                              hipStream_t stream) {
    const float*     x  = (const float*)d_in[0];
    const long long* ei = (const long long*)d_in[1];   // int64 [2,E]
    const float* W_in  = (const float*)d_in[2];
    const float* b_in  = (const float*)d_in[3];
    const float* W1    = (const float*)d_in[4];
    const float* as1   = (const float*)d_in[5];
    const float* ad1   = (const float*)d_in[6];
    const float* bias1 = (const float*)d_in[7];
    const float* W2    = (const float*)d_in[8];
    const float* as2   = (const float*)d_in[9];
    const float* ad2   = (const float*)d_in[10];
    const float* bias2 = (const float*)d_in[11];
    const float* W_se  = (const float*)d_in[12];
    const float* b_se  = (const float*)d_in[13];
    const float* W_a1  = (const float*)d_in[14];
    const float* b_a1  = (const float*)d_in[15];
    const float* W_a2  = (const float*)d_in[16];
    const float* b_a2  = (const float*)d_in[17];

    const int Nn = in_sizes[0] / 64;     // 30000
    const int E  = in_sizes[1] / 2;      // 480000
    const long long Etot = (long long)E + Nn;

    // Workspace partition (256B aligned cursor)
    char* cur = (char*)d_ws;
    auto alloc = [&](size_t bytes) -> void* {
        void* p = cur; cur += (bytes + 255) & ~(size_t)255; return p;
    };
    float*     buf0  = (float*)alloc((size_t)Nn * HID * 4);
    float*     buf1  = (float*)alloc((size_t)Nn * HID * 4);
    float*     buf2  = (float*)alloc((size_t)Nn * HID * 4);
    float*     asrcB = (float*)alloc((size_t)Nn * HEADS * 4);
    float*     adstB = (float*)alloc((size_t)Nn * HEADS * 4);
    unsigned*  mval  = (unsigned*)alloc((size_t)Nn * HEADS * 4);
    float*     denom = (float*)alloc((size_t)Nn * HEADS * 4);
    float*     exw   = (float*)alloc((size_t)Etot * HEADS * 4);
    float*     bufA1 = (float*)alloc((size_t)Nn * 64 * 4);
    _Float16*  pWin  = (_Float16*)alloc((size_t)64 * 128 * 2);
    _Float16*  pW1   = (_Float16*)alloc((size_t)128 * 128 * 2);
    _Float16*  pW2   = (_Float16*)alloc((size_t)128 * 128 * 2);
    _Float16*  pWse  = (_Float16*)alloc((size_t)128 * 128 * 2);
    _Float16*  pWa1  = (_Float16*)alloc((size_t)128 * 64 * 2);
    _Float16*  pWa2  = (_Float16*)alloc((size_t)64 * 32 * 2);
    float*     zerob = (float*)alloc(HID * 4);
    if ((size_t)(cur - (char*)d_ws) > ws_size) return;   // ws too small

    hipMemsetAsync(zerob, 0, HID * 4, stream);

    // ---- pack weights into f16 WMMA fragments (L2-resident, ~135KB) ----
    auto pack = [&](const float* W, _Float16* Wp, int K, int nout) {
        int total = (K >> 5) * (nout >> 4) * 32;
        pack_b_k<<<(total + 255) / 256, 256, 0, stream>>>(W, Wp, K, nout);
    };
    pack(W_in, pWin, 64, 128);
    pack(W1,   pW1, 128, 128);
    pack(W2,   pW2, 128, 128);
    pack(W_se, pWse, 128, 128);
    pack(W_a1, pWa1, 128, 64);
    pack(W_a2, pWa2, 64, 32);

    const int gb  = (Nn / 16 + 7) / 8;               // 8 waves/block
    const int eb  = (int)((Etot + 255) / 256);
    const int eab = (int)((Etot * HID + 255) / 256);
    const size_t nodeTot = (size_t)Nn * HID;
    const int nb = (int)((nodeTot + 255) / 256);

    // ---- h0 = relu(x @ W_in + b_in) ----
    gemm_wmma<64, 128, 1><<<gb, 256, 0, stream>>>(x, pWin, b_in, buf0, Nn);

    // ---- GAT layer 1: buf0 -> buf2 (relu'd) ----
    gemm_wmma<128, 128, 0><<<gb, 256, 0, stream>>>(buf0, pW1, zerob, buf1, Nn);
    att_scores_k<<<Nn, 128, 0, stream>>>(buf1, as1, ad1, asrcB, adstB);
    hipMemsetAsync(mval, 0, (size_t)Nn * HEADS * 4, stream);
    hipMemsetAsync(denom, 0, (size_t)Nn * HEADS * 4, stream);
    edge_max_k<<<eb, 256, 0, stream>>>(ei, E, Nn, asrcB, adstB, mval);
    edge_expsum_k<<<eb, 256, 0, stream>>>(ei, E, Nn, asrcB, adstB, mval, exw, denom);
    seed_bias_k<<<nb, 256, 0, stream>>>(buf2, bias1, nodeTot);
    edge_agg_k<<<eab, 256, 0, stream>>>(ei, E, Nn, buf1, exw, denom, buf2);
    relu_inplace_k<<<nb, 256, 0, stream>>>(buf2, nodeTot);

    // ---- GAT layer 2: buf2 -> buf1 (relu'd) ----
    gemm_wmma<128, 128, 0><<<gb, 256, 0, stream>>>(buf2, pW2, zerob, buf0, Nn);
    att_scores_k<<<Nn, 128, 0, stream>>>(buf0, as2, ad2, asrcB, adstB);
    hipMemsetAsync(mval, 0, (size_t)Nn * HEADS * 4, stream);
    hipMemsetAsync(denom, 0, (size_t)Nn * HEADS * 4, stream);
    edge_max_k<<<eb, 256, 0, stream>>>(ei, E, Nn, asrcB, adstB, mval);
    edge_expsum_k<<<eb, 256, 0, stream>>>(ei, E, Nn, asrcB, adstB, mval, exw, denom);
    seed_bias_k<<<nb, 256, 0, stream>>>(buf1, bias2, nodeTot);
    edge_agg_k<<<eab, 256, 0, stream>>>(ei, E, Nn, buf0, exw, denom, buf1);
    relu_inplace_k<<<nb, 256, 0, stream>>>(buf1, nodeTot);

    // ---- state encoder + action head ----
    gemm_wmma<128, 128, 1><<<gb, 256, 0, stream>>>(buf1, pWse, b_se, buf0, Nn);
    gemm_wmma<128, 64, 1><<<gb, 256, 0, stream>>>(buf0, pWa1, b_a1, bufA1, Nn);
    gemm_wmma<64, 32, 0><<<gb, 256, 0, stream>>>(bufA1, pWa2, b_a2, (float*)d_out, Nn);
}
